// Mamba3Triton_55241869361708
// MI455X (gfx1250) — matrix operand order repaired and checked by reading, not compiled, not run on hardware
//
#include <hip/hip_runtime.h>
#include <hip/hip_bf16.h>

typedef _Float16 h8  __attribute__((ext_vector_type(8)));
typedef _Float16 h16 __attribute__((ext_vector_type(16)));
typedef float    f8  __attribute__((ext_vector_type(8)));

#define B_  2
#define L_  4096
#define H_  24
#define P_  64
#define N_  128
#define CS_ 128
#define NC_ (L_ / CS_)

// ---- WMMA fragment loaders (CDNA5 16-bit layouts, wave32) ----
__device__ __forceinline__ h16 load_afrag(const _Float16* base, int m0, int k0, int lane) {
    const int row  = m0 + (lane & 15);
    const int half = (lane >> 4) & 1;
    const _Float16* p = base + row * 128 + k0 + half * 16;
    return *(const h16*)(p);
}

__device__ __forceinline__ h16 load_bfragT(const _Float16* baseT, int n0, int k0, int lane) {
    const int col  = n0 + (lane & 15);
    const int half = (lane >> 4) & 1;
    return *(const h16*)(baseT + col * 128 + k0 + half * 16);
}

__device__ __forceinline__ f8 wmma_f16(h16 a, h16 b, f8 c) {
    return __builtin_amdgcn_wmma_f32_16x16x32_f16(false, a, false, b, (short)0, c, false, false);
}

// Async global->LDS copy of 128 contiguous bytes (8x B128), per-lane private region.
// INST_OFFSET is added to BOTH the LDS and global addresses per the ISA pseudocode.
__device__ __forceinline__ void async_copy_128B(const float* gsrc, float* lds_dst) {
    unsigned ldsa = (unsigned)(uintptr_t)lds_dst;   // generic LDS addr: low 32 bits = LDS offset
    asm volatile("global_load_async_to_lds_b128 %0, %1, off offset:0"   :: "v"(ldsa), "v"(gsrc) : "memory");
    asm volatile("global_load_async_to_lds_b128 %0, %1, off offset:16"  :: "v"(ldsa), "v"(gsrc) : "memory");
    asm volatile("global_load_async_to_lds_b128 %0, %1, off offset:32"  :: "v"(ldsa), "v"(gsrc) : "memory");
    asm volatile("global_load_async_to_lds_b128 %0, %1, off offset:48"  :: "v"(ldsa), "v"(gsrc) : "memory");
    asm volatile("global_load_async_to_lds_b128 %0, %1, off offset:64"  :: "v"(ldsa), "v"(gsrc) : "memory");
    asm volatile("global_load_async_to_lds_b128 %0, %1, off offset:80"  :: "v"(ldsa), "v"(gsrc) : "memory");
    asm volatile("global_load_async_to_lds_b128 %0, %1, off offset:96"  :: "v"(ldsa), "v"(gsrc) : "memory");
    asm volatile("global_load_async_to_lds_b128 %0, %1, off offset:112" :: "v"(ldsa), "v"(gsrc) : "memory");
}

__global__ __launch_bounds__(256) void mamba3_fused(
    const float* __restrict__ X,  const float* __restrict__ DT,
    const float* __restrict__ Av, const float* __restrict__ Bt,
    const float* __restrict__ Ct, const float* __restrict__ Dv,
    float* __restrict__ Y)
{
    // LDS staging (~200 KB of the 320 KB WGP pool)
    __shared__ __align__(32) _Float16 sQ  [CS_ * N_];   // rotated q, [t][n]
    __shared__ __align__(32) _Float16 sKP [CS_ * N_];   // rotated k [s][n], later P [t][s]
    __shared__ __align__(32) _Float16 sKdT[N_ * CS_];   // kd^T, [n][s]
    __shared__ __align__(32) _Float16 sVt [P_ * CS_];   // v^T,  [p][s]
    __shared__ __align__(32) _Float16 sStT[P_ * N_];    // prev-state^T f16, [p][n]
    __shared__ float sSf[P_ * N_];                      // running state f32, [p][n]
    __shared__ __align__(16) float sXstage[CS_ * P_];   // async x staging (f32, next chunk)
    __shared__ float sDt[CS_], sCdt[CS_], sCum[CS_], sECum[CS_], sERD[CS_],
                     sKdS[CS_], sCos[CS_], sSin[CS_];
    __shared__ float sAng;

    const int tid  = threadIdx.x;
    const int lane = tid & 31;
    const int w    = tid >> 5;          // 8 waves
    const int bh   = blockIdx.x;
    const int bb   = bh / H_;
    const int hh   = bh % H_;
    const float Ah = Av[hh];
    const float Dh = Dv[hh];

    // per-thread private x region: half a row (32 floats = 128 B contiguous in global)
    const int xt = tid >> 1;            // row t within chunk
    const int xj = (tid & 1) * 32;      // col start
    float* myStage = &sXstage[tid * 32];

    for (int i = tid; i < P_ * N_; i += 256) sSf[i] = 0.f;
    if (tid == 0) sAng = 0.f;

    // preload chunk 0's x tile asynchronously
    {
        const float* Xp0 = X + ((size_t)(bb * L_) * H_ + hh) * P_;
        async_copy_128B(Xp0 + (size_t)xt * H_ * P_ + xj, myStage);
    }

    for (int c = 0; c < NC_; ++c) {
        __syncthreads();
        const int t0 = c * CS_;

        // ---- dt load + inclusive scan (Hillis-Steele) ----
        if (tid < CS_) {
            float d = DT[(size_t)(bb * L_ + t0 + tid) * H_ + hh];
            sDt[tid] = d; sCdt[tid] = d;
        }
        __syncthreads();
        for (int off = 1; off < CS_; off <<= 1) {
            float add = 0.f;
            if (tid < CS_ && tid >= off) add = sCdt[tid - off];
            __syncthreads();
            if (tid < CS_) sCdt[tid] += add;
            __syncthreads();
        }
        if (tid < CS_) {
            float cum = Ah * sCdt[tid];
            sCum[tid]  = cum;
            sECum[tid] = __expf(cum);
            sERD[tid]  = __expf(-cum) * sDt[tid];
            float ang  = sAng + sCdt[tid];
            sCos[tid]  = __cosf(ang);
            sSin[tid]  = __sinf(ang);
        }
        __syncthreads();
        if (tid < CS_) sKdS[tid] = __expf(sCum[CS_ - 1] - sCum[tid]) * sDt[tid];
        if (tid == 0)  sAng += sCdt[CS_ - 1];
        __syncthreads();

        // ---- x tile: wait for async copy, convert+transpose, then prefetch next ----
        {
            asm volatile("s_wait_asynccnt 0x0" ::: "memory");
            #pragma unroll
            for (int m = 0; m < 32; ++m) {
                sVt[(xj + m) * CS_ + xt] = (_Float16)myStage[m];
            }
            // ensure our staged reads (and sVt writes) are done before async overwrites
            asm volatile("s_wait_dscnt 0x0" ::: "memory");
            if (c + 1 < NC_) {
                const float* Xpn = X + ((size_t)(bb * L_ + t0 + CS_) * H_ + hh) * P_;
                async_copy_128B(Xpn + (size_t)xt * H_ * P_ + xj, myStage);
                // pull next chunk's B/C tiles toward L2/L0 (256 B per thread covers 64 KB each)
                __builtin_prefetch(Ct + (size_t)(bb * L_ + t0 + CS_) * N_ + tid * 64, 0, 0);
                __builtin_prefetch(Bt + (size_t)(bb * L_ + t0 + CS_) * N_ + tid * 64, 0, 0);
            }
        }

        // ---- RoPE q/k, kd^T, state^T staging ----
        {
            const float* Cp = Ct + (size_t)(bb * L_ + t0) * N_;
            const float* Bp = Bt + (size_t)(bb * L_ + t0) * N_;
            for (int i = tid; i < CS_ * (N_ / 2); i += 256) {
                int t = i >> 6, j = i & 63;
                float cv = sCos[t], sv = sSin[t];
                float c1 = Cp[t * N_ + j], c2 = Cp[t * N_ + j + 64];
                sQ[t * N_ + j]      = (_Float16)(c1 * cv - c2 * sv);
                sQ[t * N_ + j + 64] = (_Float16)(c1 * sv + c2 * cv);
                float b1 = Bp[t * N_ + j], b2 = Bp[t * N_ + j + 64];
                float ka = b1 * cv - b2 * sv;
                float kb = b1 * sv + b2 * cv;
                sKP[t * N_ + j]      = (_Float16)ka;
                sKP[t * N_ + j + 64] = (_Float16)kb;
                float kd = sKdS[t];
                sKdT[j * CS_ + t]        = (_Float16)(ka * kd);
                sKdT[(j + 64) * CS_ + t] = (_Float16)(kb * kd);
            }
            for (int i = tid; i < P_ * N_; i += 256) sStT[i] = (_Float16)sSf[i];
        }
        __syncthreads();

        // ---- GEMM1: scores[t][s] = sum_n q[t][n] k[s][n] ----
        f8 accS[8] = {};
        #pragma unroll
        for (int k = 0; k < 4; ++k) {
            h16 a = load_afrag(sQ, w * 16, k * 32, lane);
            #pragma unroll
            for (int nt = 0; nt < 8; ++nt) {
                h16 b = load_bfragT(sKP, nt * 16, k * 32, lane);
                accS[nt] = wmma_f16(a, b, accS[nt]);
            }
        }
        __syncthreads();   // everyone done reading sKP as K

        // ---- mask/decay/dt scale -> P, stored over sKP ----
        {
            const int half = (lane >> 4) & 1;
            #pragma unroll
            for (int nt = 0; nt < 8; ++nt) {
                #pragma unroll
                for (int r = 0; r < 8; ++r) {
                    int t = w * 16 + half * 8 + r;
                    int s = nt * 16 + (lane & 15);
                    float v = (s <= t) ? accS[nt][r] * sECum[t] * sERD[s] : 0.f;
                    sKP[t * N_ + s] = (_Float16)v;
                }
            }
        }
        __syncthreads();

        // ---- GEMM2 (P.v) and GEMM4 (q.prev) fused over same output tile ----
        f8 accY[4] = {}, accI[4] = {};
        #pragma unroll
        for (int k = 0; k < 4; ++k) {
            h16 aP = load_afrag(sKP, w * 16, k * 32, lane);
            h16 aQ = load_afrag(sQ,  w * 16, k * 32, lane);
            #pragma unroll
            for (int nt = 0; nt < 4; ++nt) {
                h16 bV = load_bfragT(sVt,  nt * 16, k * 32, lane);
                h16 bS = load_bfragT(sStT, nt * 16, k * 32, lane);
                accY[nt] = wmma_f16(aP, bV, accY[nt]);
                accI[nt] = wmma_f16(aQ, bS, accI[nt]);
            }
        }
        // ---- epilogue: y = y_intra + e^cum[t]*y_inter + v*D ----
        {
            const int half = (lane >> 4) & 1;
            #pragma unroll
            for (int nt = 0; nt < 4; ++nt) {
                #pragma unroll
                for (int r = 0; r < 8; ++r) {
                    int t = w * 16 + half * 8 + r;
                    int p = nt * 16 + (lane & 15);
                    float y = accY[nt][r] + sECum[t] * accI[nt][r]
                            + (float)sVt[p * CS_ + t] * Dh;
                    Y[((size_t)(bb * L_ + t0 + t) * H_ + hh) * P_ + p] = y;
                }
            }
        }

        // ---- GEMM3: chunk_states[n][p] = sum_s kd[s][n] v[s][p] ----
        f8 accU[4] = {};
        #pragma unroll
        for (int k = 0; k < 4; ++k) {
            h16 aK = load_afrag(sKdT, w * 16, k * 32, lane);
            #pragma unroll
            for (int nt = 0; nt < 4; ++nt) {
                h16 bV = load_bfragT(sVt, nt * 16, k * 32, lane);
                accU[nt] = wmma_f16(aK, bV, accU[nt]);
            }
        }
        // ---- state update: S = S*chunk_decay + states (f32 in LDS) ----
        {
            const float cdk = __expf(sCum[CS_ - 1]);
            const int half = (lane >> 4) & 1;
            #pragma unroll
            for (int nt = 0; nt < 4; ++nt) {
                #pragma unroll
                for (int r = 0; r < 8; ++r) {
                    int nn = w * 16 + half * 8 + r;
                    int p  = nt * 16 + (lane & 15);
                    sSf[p * N_ + nn] = sSf[p * N_ + nn] * cdk + accU[nt][r];
                }
            }
        }
    }
}

extern "C" void kernel_launch(void* const* d_in, const int* in_sizes, int n_in,
                              void* d_out, int out_size, void* d_ws, size_t ws_size,
                              hipStream_t stream) {
    const float* X  = (const float*)d_in[0];
    const float* DT = (const float*)d_in[1];
    const float* Av = (const float*)d_in[2];
    const float* Bt = (const float*)d_in[3];
    const float* Ct = (const float*)d_in[4];
    const float* Dv = (const float*)d_in[5];
    float* Y = (float*)d_out;
    dim3 grid(B_ * H_), block(256);
    hipLaunchKernelGGL(mamba3_fused, grid, block, 0, stream, X, DT, Av, Bt, Ct, Dv, Y);
}
